// SliceLSTM_30983894073947
// MI455X (gfx1250) — compile-verified, matrix operand order checked by reading, not run on hardware
//
#include <hip/hip_runtime.h>

// Problem constants (match reference)
#define B_    128
#define S_    512
#define NSL_  4
#define INSZ_ 128
#define HID_  128
#define H_    512      // NSL*HID
#define GW_   512      // 4*HID per slice (gate width)
#define K1_   256      // INSZ + HID (fused x|h K for stage-1 GEMM)

typedef _Float16 hv8  __attribute__((ext_vector_type(8)));
typedef _Float16 hv16 __attribute__((ext_vector_type(16)));
typedef float    fv8  __attribute__((ext_vector_type(8)));

static __device__ __forceinline__ hv16 pack16(hv8 lo, hv8 hi) {
    hv16 r;
#pragma unroll
    for (int i = 0; i < 8; ++i) { r[i] = lo[i]; r[i + 8] = hi[i]; }
    return r;
}
static __device__ __forceinline__ float sigf(float x) {
    return 1.0f / (1.0f + __expf(-x));
}

// ---------------------------------------------------------------------------
// One-time weight conversion: w1t[s][n][k] (f16), n in [0,512), k in [0,256):
//   k <  128 -> Ws[s][k][n]       (input weights)
//   k >= 128 -> Us[s][k-128][n]   (recurrent weights)
// Transposed so WMMA B-fragments load with the same addressing as A-fragments.
// ---------------------------------------------------------------------------
__global__ __launch_bounds__(256) void convW1_kernel(
    const float* __restrict__ Ws, const float* __restrict__ Us,
    _Float16* __restrict__ w1t)
{
    int idx = blockIdx.x * 256 + threadIdx.x;          // 4*512*256 = 524288
    if (idx >= NSL_ * GW_ * K1_) return;
    int k = idx & 255;
    int n = (idx >> 8) & 511;
    int s = idx >> 17;
    float v = (k < 128) ? Ws[((size_t)s * INSZ_ + k) * GW_ + n]
                        : Us[((size_t)s * HID_ + (k - 128)) * GW_ + n];
    w1t[idx] = (_Float16)v;
}

// wct[q][n][k] (f16) = connector_W[k][q*512+n]
__global__ __launch_bounds__(256) void convW2_kernel(
    const float* __restrict__ Wc, _Float16* __restrict__ wct)
{
    int idx = blockIdx.x * 256 + threadIdx.x;          // 4*512*512 = 1048576
    if (idx >= 4 * H_ * H_) return;
    int k = idx & 511;
    int n = (idx >> 9) & 511;
    int q = idx >> 18;
    wct[idx] = (_Float16)Wc[(size_t)k * (4 * H_) + q * H_ + n];
}

__global__ __launch_bounds__(256) void init_state_kernel(
    float* __restrict__ cws, _Float16* __restrict__ h16b)
{
    int i = blockIdx.x * 256 + threadIdx.x;
    if (i < B_ * H_) { cws[i] = 0.0f; h16b[i] = (_Float16)0.0f; }
}

// ---------------------------------------------------------------------------
// Stage 1 (per step): per-slice gates = [x_t | h] @ [Ws;Us] + bias, activated.
// Output matrix: M=128 (batch), N=2048 (= slice*512 + gate*128 + unit).
// Grid: 128 blocks of 256 thr. block -> mt (M tile 0..7), ng (N group 0..15).
// Each ng spans 128 columns => one slice s = ng>>2, one gate quadrant q = ng&3.
// 8 waves per block, each one 16x16 tile, K=256 -> 8 WMMA per wave.
// A tile (16 x 256 f16) staged in LDS: x fp32->f16 convert + h f16 copy.
// ---------------------------------------------------------------------------
__global__ __launch_bounds__(256) void gates_kernel(
    const float* __restrict__ x, const _Float16* __restrict__ w1t,
    const float* __restrict__ biases, const _Float16* __restrict__ hprev,
    _Float16* __restrict__ actI, _Float16* __restrict__ actF,
    _Float16* __restrict__ actG, _Float16* __restrict__ actO, int t)
{
    __shared__ _Float16 As[16][264];                   // +8 halfs pad (bank spread)
    const int tid = threadIdx.x;
    const int mt  = blockIdx.x & 7;
    const int ng  = blockIdx.x >> 3;
    const int m0  = mt * 16;
    const int s   = ng >> 2;                           // slice
    const int q   = ng & 3;                            // gate quadrant (i,f,g,o)

    {   // cooperative A fill: 16 rows x 256 K
        const int row = tid >> 4;
        const int c16 = (tid & 15) * 16;
        const int b   = m0 + row;
        if (c16 < 128) {                               // x part, fp32 -> f16
            const float* xp = x + ((size_t)b * S_ + t) * H_ + s * 128 + c16;
#pragma unroll
            for (int j = 0; j < 16; ++j) As[row][c16 + j] = (_Float16)xp[j];
        } else {                                       // h part, f16 copy
            const _Float16* hp = hprev + (size_t)b * H_ + s * 128 + (c16 - 128);
            *(hv8*)&As[row][c16]     = *(const hv8*)hp;
            *(hv8*)&As[row][c16 + 8] = *(const hv8*)(hp + 8);
        }
    }
    __syncthreads();

    const int w    = tid >> 5;
    const int lane = tid & 31;
    const int hi   = lane >> 4;
    const int ln   = lane & 15;
    const int g0   = q * 128 + w * 16;                 // column within slice
    const _Float16* bp = w1t + ((size_t)(s * GW_ + g0 + ln)) * K1_;

    fv8 acc = {};
#pragma unroll
    for (int kk = 0; kk < 8; ++kk) {
        const int kA = kk * 32 + hi * 8;
        hv16 a  = pack16(*(const hv8*)&As[ln][kA], *(const hv8*)&As[ln][kA + 16]);
        hv16 bm = pack16(*(const hv8*)(bp + kA),   *(const hv8*)(bp + kA + 16));
        acc = __builtin_amdgcn_wmma_f32_16x16x32_f16(
            false, a, false, bm, (short)0, acc, false, false);
    }

    const float bias = biases[s * GW_ + g0 + ln];
    _Float16* dst = (q == 0) ? actI : (q == 1) ? actF : (q == 2) ? actG : actO;
    const int hcol = s * 128 + w * 16 + ln;            // column in [0, H)
#pragma unroll
    for (int r = 0; r < 8; ++r) {
        const int m = r + hi * 8;
        float v = acc[r] + bias;
        v = (q == 2) ? tanhf(v) : sigf(v);
        dst[(size_t)(m0 + m) * H_ + hcol] = (_Float16)v;
    }
}

// ---------------------------------------------------------------------------
// Stage 2 (per step): 4 connector GEMMs (M=128,K=512,N=512) + fused cell update.
// Grid: 128 blocks of 256 thr. block -> mt (0..7), np (0..15) = 2 N tiles.
// 8 waves: wave w handles gate q = w&3 of tile ts = w>>2 (16 WMMA, K=512).
// z deposited in LDS; after barrier the block fuses c/h update and writes
// hidden_seq (fp32), cell state (fp32) and h (f16, next-step WMMA operand).
// ---------------------------------------------------------------------------
__global__ __launch_bounds__(256) void connector_kernel(
    const _Float16* __restrict__ actI, const _Float16* __restrict__ actF,
    const _Float16* __restrict__ actG, const _Float16* __restrict__ actO,
    const _Float16* __restrict__ wct, const float* __restrict__ cb,
    float* __restrict__ cws, _Float16* __restrict__ hnext,
    float* __restrict__ out, int t)
{
    __shared__ float zbuf[8][16][17];                  // +1 pad
    const int tid  = threadIdx.x;
    const int mt   = blockIdx.x >> 4;                  // 0..7
    const int np   = blockIdx.x & 15;                  // 0..15
    const int m0   = mt * 16;
    const int w    = tid >> 5;
    const int lane = tid & 31;
    const int hi   = lane >> 4;
    const int ln   = lane & 15;
    const int q    = w & 3;                            // which gate GEMM
    const int ts   = w >> 2;                           // which of 2 N tiles
    const int n0   = np * 32 + ts * 16;

    const _Float16* act = (q == 0) ? actI : (q == 1) ? actF : (q == 2) ? actG : actO;
    const _Float16* ap  = act + (size_t)(m0 + ln) * H_;
    const _Float16* bp  = wct + ((size_t)(q * H_ + n0 + ln)) * H_;

    fv8 acc = {};
#pragma unroll
    for (int kk = 0; kk < 16; ++kk) {
        const int kA = kk * 32 + hi * 8;
        hv16 a  = pack16(*(const hv8*)(ap + kA), *(const hv8*)(ap + kA + 16));
        hv16 bm = pack16(*(const hv8*)(bp + kA), *(const hv8*)(bp + kA + 16));
        acc = __builtin_amdgcn_wmma_f32_16x16x32_f16(
            false, a, false, bm, (short)0, acc, false, false);
    }

    const float bias = cb[q * H_ + n0 + ln];
#pragma unroll
    for (int r = 0; r < 8; ++r)
        zbuf[w][r + hi * 8][ln] = acc[r] + bias;
    __syncthreads();

    // fused LSTM cell update: 2 tiles * 256 elems, 256 threads -> 2 each
#pragma unroll
    for (int e = 0; e < 2; ++e) {
        const int idx  = e * 256 + tid;
        const int tile = idx >> 8;
        const int m    = (idx >> 4) & 15;
        const int n    = idx & 15;
        const int wb   = tile * 4;
        const float z0 = zbuf[wb + 0][m][n];
        const float z1 = zbuf[wb + 1][m][n];
        const float z2 = zbuf[wb + 2][m][n];
        const float z3 = zbuf[wb + 3][m][n];
        const float i2 = sigf(z0);
        const float f2 = sigf(z1);
        const float g2 = tanhf(z2);
        const float o2 = sigf(z3);
        const int b    = m0 + m;
        const int col  = np * 32 + tile * 16 + n;
        const size_t ci = (size_t)b * H_ + col;
        const float cn = f2 * cws[ci] + i2 * g2;
        const float h  = o2 * tanhf(cn);
        cws[ci] = cn;
        out[((size_t)b * S_ + t) * H_ + col] = h;
        hnext[ci] = (_Float16)h;
    }
}

// Copy last h and c into the h_t / c_t output regions
__global__ __launch_bounds__(256) void finalize_kernel(
    float* __restrict__ out, const float* __restrict__ cws)
{
    int i = blockIdx.x * 256 + threadIdx.x;
    if (i >= B_ * H_) return;
    const size_t hseq = (size_t)B_ * S_ * H_;
    int b = i >> 9, j = i & 511;
    out[hseq + i] = out[((size_t)b * S_ + (S_ - 1)) * H_ + j];
    out[hseq + (size_t)B_ * H_ + i] = cws[i];
}

extern "C" void kernel_launch(void* const* d_in, const int* in_sizes, int n_in,
                              void* d_out, int out_size, void* d_ws, size_t ws_size,
                              hipStream_t stream) {
    const float* x  = (const float*)d_in[0];   // [B,S,512]
    const float* Ws = (const float*)d_in[1];   // [4,128,512]
    const float* Us = (const float*)d_in[2];   // [4,128,512]
    const float* bi = (const float*)d_in[3];   // [4,512]
    const float* Wc = (const float*)d_in[4];   // [512,2048]
    const float* cb = (const float*)d_in[5];   // [2048]
    float* out = (float*)d_out;                // hidden_seq | h_t | c_t

    // workspace layout (all offsets 16B-aligned; total < 4 MB)
    char* ws = (char*)d_ws;
    _Float16* w1t  = (_Float16*)(ws + 0);               // 1,048,576 B
    _Float16* wct  = (_Float16*)(ws + (1u << 20));      // 2,097,152 B
    _Float16* actI = (_Float16*)(ws + 3u * (1u << 20)); // 131,072 B each
    _Float16* actF = (_Float16*)(ws + 3u * (1u << 20) + 1u * 131072u);
    _Float16* actG = (_Float16*)(ws + 3u * (1u << 20) + 2u * 131072u);
    _Float16* actO = (_Float16*)(ws + 3u * (1u << 20) + 3u * 131072u);
    _Float16* h16b = (_Float16*)(ws + 3u * (1u << 20) + 4u * 131072u);
    float*    cws  = (float*)   (ws + 3u * (1u << 20) + 5u * 131072u); // 262,144 B

    convW1_kernel<<<(NSL_ * GW_ * K1_) / 256, 256, 0, stream>>>(Ws, Us, w1t);
    convW2_kernel<<<(4 * H_ * H_) / 256, 256, 0, stream>>>(Wc, wct);
    init_state_kernel<<<(B_ * H_) / 256, 256, 0, stream>>>(cws, h16b);

    for (int t = 0; t < S_; ++t) {
        gates_kernel<<<128, 256, 0, stream>>>(x, w1t, bi, h16b,
                                              actI, actF, actG, actO, t);
        connector_kernel<<<128, 256, 0, stream>>>(actI, actF, actG, actO,
                                                  wct, cb, cws, h16b, out, t);
    }

    finalize_kernel<<<(B_ * H_) / 256, 256, 0, stream>>>(out, cws);
}
